// DecoderLayer_12369505812685
// MI455X (gfx1250) — matrix operand order repaired and checked by reading, not compiled, not run on hardware
//
#include <hip/hip_runtime.h>

// ---------------------------------------------------------------------------
// DecoderLayer for MI455X (gfx1250), compile-only tuned:
//  * all matrix math through v_wmma_f32_16x16x32_f16 (f32 accumulate)
//  * X and all weights pre-converted to f16 once -> GEMM hot loop is pure
//    b128 loads + WMMA (no v_cvt in the k-loop), halves streamed bytes
//  * flash attention computes S^T = K x Q^T so the exp'd score tile IS the
//    A-fragment of P x V (zero cross-lane movement); log2-domain softmax
//    (log2e folded into Q scale) -> raw v_exp_f32 via __builtin_amdgcn_exp2f
//  * causal mask only in the single edge block per query tile (no cndmask
//    chains in the steady-state loop)
// ---------------------------------------------------------------------------

typedef __attribute__((ext_vector_type(16))) _Float16 v16h;
typedef __attribute__((ext_vector_type(8)))  _Float16 v8h;
typedef __attribute__((ext_vector_type(8)))  float    v8f;

#define WMMA_F16(a, b, c) \
  __builtin_amdgcn_wmma_f32_16x16x32_f16(false, (a), false, (b), (short)0, (c), false, false)

// raw v_exp_f32: no libm denormal fixup; args are <= 0 and flush-to-zero for
// very negative inputs is exactly what softmax wants.
#define FAST_EXP2(x) __builtin_amdgcn_exp2f(x)

static constexpr int S_LEN  = 2048;
static constexpr int DMODEL = 1024;
static constexpr int NHEADS = 16;
static constexpr int DK     = 64;
static constexpr float LOG2E = 1.44269504088896340736f;

// ---------------- fragment loaders (contiguous k order, same for A and B) -------

__device__ __forceinline__ v16h cat8(v8h lo, v8h hi) {
  v16h r;
#pragma unroll
  for (int i = 0; i < 8; ++i) { r[i] = lo[i]; r[8 + i] = hi[i]; }
  return r;
}

// A 16x32 fragment from f16 row-major: two contiguous 16B groups.
// lane<16: row, k = k0..k0+15; lane>=16: k = k0+16..k0+31
__device__ __forceinline__ v16h load_a_f16(const _Float16* __restrict__ base, int ld,
                                           int row, int k0, int lane) {
  const _Float16* p = base + (size_t)row * ld + k0 + ((lane & 16) ? 16 : 0);
  return cat8(*(const v8h*)p, *(const v8h*)(p + 8));
}

// B 32x16 fragment where B[k,n] = W[n,k], W f16 row-major: one 32B group.
// lane<16: col n, k = k0..k0+15 ; lane>=16: k = k0+16..k0+31
__device__ __forceinline__ v16h load_b_f16(const _Float16* __restrict__ base, int ld,
                                           int col, int k0, int lane) {
  return *(const v16h*)(base + (size_t)col * ld + k0 + ((lane & 16) ? 16 : 0));
}

// sinusoidal positional embedding value
__device__ __forceinline__ float pe_val(int s, int d) {
  const float div = __powf(10000.0f, (float)(2 * (d >> 1)) * (1.0f / 64.0f));
  const float ang = (float)s / div;
  return (d & 1) ? __cosf(ang) : __sinf(ang);
}

// ---------------- f32 -> f16 pre-conversion ---------------------------------
__global__ void __launch_bounds__(256)
cvt_f16_kernel(const float* __restrict__ src, _Float16* __restrict__ dst) {
  const size_t i = ((size_t)blockIdx.x * 256 + threadIdx.x) * 8;
  const float4 a = ((const float4*)(src + i))[0];
  const float4 b = ((const float4*)(src + i))[1];
  v8h o;
  o[0] = (_Float16)a.x; o[1] = (_Float16)a.y; o[2] = (_Float16)a.z; o[3] = (_Float16)a.w;
  o[4] = (_Float16)b.x; o[5] = (_Float16)b.y; o[6] = (_Float16)b.z; o[7] = (_Float16)b.w;
  *(v8h*)(dst + i) = o;
}

// ---------------- QKV projection GEMMs -------------------------------------
// C[m,n] = sum_k X[m,k] * W[n,k], f16 inputs, f32 accum.
// Per-wave 64x64 tile (4x4 WMMA accumulators): 16 b128 loads : 16 WMMA.
// MODE 0 (Q): epilogue adds PE, folds 0.125*log2e, stores f16 [b,h,s,d]
// MODE 1 (K): stores raw f32 [b,h,s,d] (smear gate needs neighbors)
// MODE 2 (V): stores f16 TRANSPOSED [b,h,d,s] via packed v8h stores
template <int MODE>
__global__ void __launch_bounds__(128)
gemm_qkv_kernel(const _Float16* __restrict__ X, const _Float16* __restrict__ W,
                _Float16* __restrict__ outH, float* __restrict__ outF) {
  const int lane = threadIdx.x & 31;
  const int wave = threadIdx.x >> 5;
  const int m0 = blockIdx.y * 128 + (wave >> 1) * 64;
  const int n0 = blockIdx.x * 128 + (wave & 1) * 64;
  const int arow = lane & 15;

  v8f c[4][4] = {};
#pragma unroll 1
  for (int k0 = 0; k0 < DMODEL; k0 += 32) {
    v16h a[4], b[4];
#pragma unroll
    for (int i = 0; i < 4; ++i) a[i] = load_a_f16(X, DMODEL, m0 + 16 * i + arow, k0, lane);
#pragma unroll
    for (int j = 0; j < 4; ++j) b[j] = load_b_f16(W, DMODEL, n0 + 16 * j + arow, k0, lane);
#pragma unroll
    for (int i = 0; i < 4; ++i)
#pragma unroll
      for (int j = 0; j < 4; ++j) c[i][j] = WMMA_F16(a[i], b[j], c[i][j]);
  }

  const int rbase = (lane & 16) ? 8 : 0;
#pragma unroll
  for (int i = 0; i < 4; ++i) {
    const int mb = m0 + 16 * i + rbase;      // first row held by this lane half
#pragma unroll
    for (int j = 0; j < 4; ++j) {
      const int n = n0 + 16 * j + (lane & 15);
      const int h = n >> 6, d = n & 63;
      if (MODE == 2) {
        const int bb = mb >> 11, s = mb & 2047;
        v8h pk;
#pragma unroll
        for (int r = 0; r < 8; ++r) pk[r] = (_Float16)c[i][j][r];
        _Float16* dst = outH + ((size_t)(bb * NHEADS + h) * DK + d) * S_LEN + s;
        *(v8h*)dst = pk;                     // 8 consecutive tokens, 16B store
      } else {
#pragma unroll
        for (int r = 0; r < 8; ++r) {
          const int m = mb + r;
          const int bb = m >> 11, s = m & 2047;
          const size_t idx = ((size_t)(bb * NHEADS + h) * S_LEN + s) * DK + d;
          const float v = c[i][j][r];
          if (MODE == 0) outH[idx] = (_Float16)((v + pe_val(s, d)) * (0.125f * LOG2E));
          else           outF[idx] = v;
        }
      }
    }
  }
}

// ---------------- K smear gate + positional embedding -----------------------
// Note: K scores get *0.125*log2e via Q's folded scale; K itself stays unscaled.
__global__ void __launch_bounds__(256)
smear_pe_kernel(const float* __restrict__ Kf, const float* __restrict__ alpha,
                _Float16* __restrict__ Kh) {
  const size_t idx = (size_t)blockIdx.x * 256 + threadIdx.x;
  const int d = (int)(idx & 63);
  const int s = (int)((idx >> 6) & 2047);
  const int h = (int)(idx >> 17) & 15;
  const float k = Kf[idx];
  float v;
  if (s == 0) {
    v = k;
  } else {
    const float a = 1.0f / (1.0f + __expf(-alpha[h * (S_LEN - 1) + (s - 1)]));
    v = a * k + (1.0f - a) * Kf[idx - 64];
  }
  Kh[idx] = (_Float16)(v + pe_val(s, d));
}

// ---------------- flash attention (causal) ----------------------------------
// One wave = 16 queries.  Per 32-wide kv block:
//   S^T tile = K_block(A) x Q^T(B)  -> C layout: kv in VGPR, q in lane.
//   All 16 score values in a lane belong to ONE query -> per-lane softmax
//   stats (+ one shfl_xor(16) to merge the two kv half-ranges).
//   exp2'd scores convert in-place into the A fragment of P x V.
// MASK=true only for the single diagonal-overlapping block per query tile.
template <bool MASK>
__device__ __forceinline__ void attn_block(
    int kv0, const _Float16* __restrict__ Kp, const _Float16* __restrict__ Vp,
    v16h qb0, v16h qb1, int lane, int col, int kh, int r8, int q,
    float& m_run, float& l_run, v8f& o0, v8f& o1, v8f& o2, v8f& o3) {
  // K block as two A fragments per 16-row kv tile (contiguous dk reads)
  const _Float16* kr0 = Kp + (size_t)(kv0 + col) * DK;
  const _Float16* kr1 = Kp + (size_t)(kv0 + 16 + col) * DK;
  const v16h ka00 = cat8(*(const v8h*)(kr0 + r8),      *(const v8h*)(kr0 + r8 + 16));
  const v16h ka01 = cat8(*(const v8h*)(kr0 + 32 + r8), *(const v8h*)(kr0 + 48 + r8));
  const v16h ka10 = cat8(*(const v8h*)(kr1 + r8),      *(const v8h*)(kr1 + r8 + 16));
  const v16h ka11 = cat8(*(const v8h*)(kr1 + 32 + r8), *(const v8h*)(kr1 + 48 + r8));

  v8f st0 = {}, st1 = {};
  st0 = WMMA_F16(ka00, qb0, st0);
  st0 = WMMA_F16(ka01, qb1, st0);
  st1 = WMMA_F16(ka10, qb0, st1);
  st1 = WMMA_F16(ka11, qb1, st1);

  float s0[8], s1[8];
#pragma unroll
  for (int r = 0; r < 8; ++r) {
    if (MASK) {
      const int kva = kv0 + r + r8;
      s0[r] = (kva > q)      ? -1.0e9f : st0[r];
      s1[r] = (kva + 16 > q) ? -1.0e9f : st1[r];
    } else {
      s0[r] = st0[r];
      s1[r] = st1[r];
    }
  }
  float bmax = -3.0e38f;
#pragma unroll
  for (int r = 0; r < 8; ++r) bmax = fmaxf(bmax, fmaxf(s0[r], s1[r]));
  bmax = fmaxf(bmax, __shfl_xor(bmax, 16, 32));
  const float mnew = fmaxf(m_run, bmax);
  const float rf = FAST_EXP2(m_run - mnew);      // log2-domain softmax
  float bsum = 0.0f;
  float p0[8], p1[8];
#pragma unroll
  for (int r = 0; r < 8; ++r) {
    p0[r] = FAST_EXP2(s0[r] - mnew);
    p1[r] = FAST_EXP2(s1[r] - mnew);
    bsum += p0[r] + p1[r];
  }
  bsum += __shfl_xor(bsum, 16, 32);
  l_run = l_run * rf + bsum;
  m_run = mnew;
  // rescale running O: per-row factor broadcast (row q = VGPR index here)
#pragma unroll
  for (int r = 0; r < 8; ++r) {
    const float f = __shfl(rf, r + ((lane & 16) >> 1), 32);
    o0[r] *= f; o1[r] *= f; o2[r] *= f; o3[r] *= f;
  }
  // P in A layout for free: tile0 regs -> K 0..7/8..15, tile1 -> K 16..31
  v16h pa;
#pragma unroll
  for (int r = 0; r < 8; ++r) { pa[r] = (_Float16)p0[r]; pa[8 + r] = (_Float16)p1[r]; }
  // V B fragments from transposed V (row = head-dim, contiguous kv)
  const v16h vb0 = cat8(*(const v8h*)(Vp + (size_t)(col)      * S_LEN + kv0 + r8), *(const v8h*)(Vp + (size_t)(col)      * S_LEN + kv0 + r8 + 16));
  const v16h vb1 = cat8(*(const v8h*)(Vp + (size_t)(16 + col) * S_LEN + kv0 + r8), *(const v8h*)(Vp + (size_t)(16 + col) * S_LEN + kv0 + r8 + 16));
  const v16h vb2 = cat8(*(const v8h*)(Vp + (size_t)(32 + col) * S_LEN + kv0 + r8), *(const v8h*)(Vp + (size_t)(32 + col) * S_LEN + kv0 + r8 + 16));
  const v16h vb3 = cat8(*(const v8h*)(Vp + (size_t)(48 + col) * S_LEN + kv0 + r8), *(const v8h*)(Vp + (size_t)(48 + col) * S_LEN + kv0 + r8 + 16));
  o0 = WMMA_F16(pa, vb0, o0);
  o1 = WMMA_F16(pa, vb1, o1);
  o2 = WMMA_F16(pa, vb2, o2);
  o3 = WMMA_F16(pa, vb3, o3);
}

__global__ void __launch_bounds__(128)
attn_kernel(const _Float16* __restrict__ Qh, const _Float16* __restrict__ Kh,
            const _Float16* __restrict__ Vt, _Float16* __restrict__ Ah) {
  __shared__ __align__(32) _Float16 lds[4][16][80];
  const int lane = threadIdx.x & 31;
  const int wave = threadIdx.x >> 5;
  const int bh = blockIdx.y;
  const int b = bh >> 4, h = bh & 15;
  const int q0 = (blockIdx.x * 4 + wave) * 16;

  const _Float16* Qp = Qh + (size_t)bh * S_LEN * DK;
  const _Float16* Kp = Kh + (size_t)bh * S_LEN * DK;
  const _Float16* Vp = Vt + (size_t)bh * DK * S_LEN;

  const int col = lane & 15;
  const int kh  = (lane & 16) ? 16 : 0;  // unused; B-fragment k offsets use r8
  const int r8  = (lane & 16) ? 8 : 0;   // A/C-fragment row-half select
  const int q   = q0 + col;              // this lane's query index

  // Q as B fragments of S^T = K x Q^T (kept in registers for all kv blocks)
  const v16h qb0 = cat8(*(const v8h*)(Qp + (size_t)(q0 + col) * DK + r8), *(const v8h*)(Qp + (size_t)(q0 + col) * DK + r8 + 16));
  const v16h qb1 = cat8(*(const v8h*)(Qp + (size_t)(q0 + col) * DK + 32 + r8), *(const v8h*)(Qp + (size_t)(q0 + col) * DK + 48 + r8));

  float m_run = -3.0e38f, l_run = 0.0f;
  v8f o0 = {}, o1 = {}, o2 = {}, o3 = {};

  // kv blocks [0, nfull) never touch the diagonal; exactly one edge block.
  const int nfull = q0 >> 5;
#pragma unroll 1
  for (int jb = 0; jb < nfull; ++jb)
    attn_block<false>(jb << 5, Kp, Vp, qb0, qb1, lane, col, kh, r8, q,
                      m_run, l_run, o0, o1, o2, o3);
  attn_block<true>(nfull << 5, Kp, Vp, qb0, qb1, lane, col, kh, r8, q,
                   m_run, l_run, o0, o1, o2, o3);

  // normalize and transpose through LDS -> row-major f16 attn [B,S,D]
  const float invl = 1.0f / l_run;
#pragma unroll
  for (int r = 0; r < 8; ++r) {
    const float f = __shfl(invl, r + ((lane & 16) >> 1), 32);
    lds[wave][r + r8][0  + col] = (_Float16)(o0[r] * f);
    lds[wave][r + r8][16 + col] = (_Float16)(o1[r] * f);
    lds[wave][r + r8][32 + col] = (_Float16)(o2[r] * f);
    lds[wave][r + r8][48 + col] = (_Float16)(o3[r] * f);
  }
  __syncthreads();
  const int rr = lane & 15;
  const int hf = lane >> 4;  // 0 or 1: which 32-wide half of the head dim
  _Float16* dst = Ah + (size_t)(b * S_LEN + q0 + rr) * DMODEL + h * DK + hf * 32;
  *(v8h*)(dst + 0)  = *(const v8h*)&lds[wave][rr][hf * 32 + 0];
  *(v8h*)(dst + 8)  = *(const v8h*)&lds[wave][rr][hf * 32 + 8];
  *(v8h*)(dst + 16) = *(const v8h*)&lds[wave][rr][hf * 32 + 16];
  *(v8h*)(dst + 24) = *(const v8h*)&lds[wave][rr][hf * 32 + 24];
}

// ---------------- output projection + residual ------------------------------
__global__ void __launch_bounds__(128)
oproj_kernel(const _Float16* __restrict__ A, const _Float16* __restrict__ W,
             const float* __restrict__ X, float* __restrict__ Y) {
  const int lane = threadIdx.x & 31;
  const int wave = threadIdx.x >> 5;
  const int m0 = blockIdx.y * 128 + (wave >> 1) * 64;
  const int n0 = blockIdx.x * 128 + (wave & 1) * 64;
  const int arow = lane & 15;

  v8f c[4][4] = {};
#pragma unroll 1
  for (int k0 = 0; k0 < DMODEL; k0 += 32) {
    v16h a[4], b[4];
#pragma unroll
    for (int i = 0; i < 4; ++i) a[i] = load_a_f16(A, DMODEL, m0 + 16 * i + arow, k0, lane);
#pragma unroll
    for (int j = 0; j < 4; ++j) b[j] = load_b_f16(W, DMODEL, n0 + 16 * j + arow, k0, lane);
#pragma unroll
    for (int i = 0; i < 4; ++i)
#pragma unroll
      for (int j = 0; j < 4; ++j) c[i][j] = WMMA_F16(a[i], b[j], c[i][j]);
  }

  const int rbase = (lane & 16) ? 8 : 0;
#pragma unroll
  for (int i = 0; i < 4; ++i) {
#pragma unroll
    for (int j = 0; j < 4; ++j) {
      const int n = n0 + 16 * j + (lane & 15);
#pragma unroll
      for (int r = 0; r < 8; ++r) {
        const size_t idx = (size_t)(m0 + 16 * i + rbase + r) * DMODEL + n;
        Y[idx] = c[i][j][r] + X[idx];  // residual
      }
    }
  }
}

// ---------------- LayerNorm --------------------------------------------------
__global__ void __launch_bounds__(256)
ln_kernel(const float* __restrict__ Y, const float* __restrict__ g,
          const float* __restrict__ bta, float* __restrict__ out) {
  const int row = blockIdx.x;
  const int tid = threadIdx.x;
  const float4 v = ((const float4*)(Y + (size_t)row * DMODEL))[tid];
  float s  = v.x + v.y + v.z + v.w;
  float ss = v.x * v.x + v.y * v.y + v.z * v.z + v.w * v.w;
#pragma unroll
  for (int m = 1; m < 32; m <<= 1) {
    s  += __shfl_xor(s, m, 32);
    ss += __shfl_xor(ss, m, 32);
  }
  __shared__ float rs[8], rss[8];
  if ((tid & 31) == 0) { rs[tid >> 5] = s; rss[tid >> 5] = ss; }
  __syncthreads();
  s = 0.0f; ss = 0.0f;
#pragma unroll
  for (int i = 0; i < 8; ++i) { s += rs[i]; ss += rss[i]; }
  const float mu  = s * (1.0f / DMODEL);
  const float var = ss * (1.0f / DMODEL) - mu * mu;
  const float inv = rsqrtf(var + 1e-5f);
  const float4 gg = ((const float4*)g)[tid];
  const float4 bb = ((const float4*)bta)[tid];
  float4 o;
  o.x = (v.x - mu) * inv * gg.x + bb.x;
  o.y = (v.y - mu) * inv * gg.y + bb.y;
  o.z = (v.z - mu) * inv * gg.z + bb.z;
  o.w = (v.w - mu) * inv * gg.w + bb.w;
  ((float4*)(out + (size_t)row * DMODEL))[tid] = o;
}

// ---------------- launcher ---------------------------------------------------
extern "C" void kernel_launch(void* const* d_in, const int* in_sizes, int n_in,
                              void* d_out, int out_size, void* d_ws, size_t ws_size,
                              hipStream_t stream) {
  const float* X     = (const float*)d_in[0];
  // d_in[1] = direct_path (unused by the reference math)
  const float* Wq    = (const float*)d_in[2];
  const float* Wk    = (const float*)d_in[3];
  const float* Wv    = (const float*)d_in[4];
  const float* Wo    = (const float*)d_in[5];
  const float* gamma = (const float*)d_in[6];
  const float* beta  = (const float*)d_in[7];
  const float* alpha = (const float*)d_in[8];

  char* ws = (char*)d_ws;
  // [0, 32M): K f32 scratch, later reused as pre-LN Y f32 (K dead by then)
  float*    Kf  = (float*)ws;
  _Float16* Qh  = (_Float16*)(ws + 33554432);   // f16 [B,H,S,dk], PE+scale*log2e
  _Float16* Kh  = (_Float16*)(ws + 50331648);   // f16 [B,H,S,dk], smear+PE
  _Float16* Vt  = (_Float16*)(ws + 67108864);   // f16 [B,H,dk,S]  (transposed)
  _Float16* Ah  = (_Float16*)(ws + 83886080);   // f16 [B,S,D] attention output
  _Float16* Xh  = (_Float16*)(ws + 100663296);  // f16 copy of X
  _Float16* Wqh = (_Float16*)(ws + 117440512);  // f16 weights
  _Float16* Wkh = (_Float16*)(ws + 119537664);
  _Float16* Wvh = (_Float16*)(ws + 121634816);
  _Float16* Woh = (_Float16*)(ws + 123731968);
  float*    Yf  = Kf;

  // one-time f32->f16 conversions (amortized across all GEMM tile re-reads)
  cvt_f16_kernel<<<(4 * S_LEN * DMODEL) / 2048, 256, 0, stream>>>(X,  Xh);
  cvt_f16_kernel<<<(DMODEL * DMODEL) / 2048, 256, 0, stream>>>(Wq, Wqh);
  cvt_f16_kernel<<<(DMODEL * DMODEL) / 2048, 256, 0, stream>>>(Wk, Wkh);
  cvt_f16_kernel<<<(DMODEL * DMODEL) / 2048, 256, 0, stream>>>(Wv, Wvh);
  cvt_f16_kernel<<<(DMODEL * DMODEL) / 2048, 256, 0, stream>>>(Wo, Woh);

  const dim3 gblk(128);
  const dim3 ggrid(DMODEL / 128, (4 * S_LEN) / 128);  // (8, 64)

  gemm_qkv_kernel<0><<<ggrid, gblk, 0, stream>>>(Xh, Wqh, Qh, nullptr);
  gemm_qkv_kernel<1><<<ggrid, gblk, 0, stream>>>(Xh, Wkh, nullptr, Kf);
  gemm_qkv_kernel<2><<<ggrid, gblk, 0, stream>>>(Xh, Wvh, Vt, nullptr);
  smear_pe_kernel<<<(4 * NHEADS * S_LEN * DK) / 256, 256, 0, stream>>>(Kf, alpha, Kh);
  attn_kernel<<<dim3(S_LEN / 64, 4 * NHEADS), 128, 0, stream>>>(Qh, Kh, Vt, Ah);
  oproj_kernel<<<ggrid, gblk, 0, stream>>>(Ah, Woh, X, Yf);
  ln_kernel<<<4 * S_LEN, 256, 0, stream>>>(Yf, gamma, beta, (float*)d_out);
}
